// RollingCorrelationGraph_19842748907740
// MI455X (gfx1250) — matrix operand order repaired and checked by reading, not compiled, not run on hardware
//
#include <hip/hip_runtime.h>
#include <hip/hip_bf16.h>
#include <cstdint>

typedef __attribute__((ext_vector_type(16))) _Float16 v16h;
typedef __attribute__((ext_vector_type(8)))  _Float16 v8h;
typedef __attribute__((ext_vector_type(4)))  _Float16 v4h;
typedef __attribute__((ext_vector_type(8)))  float    v8f;

#define LN    128
#define NN    4096
#define TOPK  20
// LDS row pitch: 128 blocks x 33 floats (+2 so pitch % 64 == 2 -> the two
// half-wave groups of a WMMA D store hit disjoint bank sets)
#define RSTR  (128 * 33 + 2)

__device__ __forceinline__ float wave_sum32(float v) {
#pragma unroll
    for (int o = 16; o > 0; o >>= 1) v += __shfl_xor(v, o, 32);
    return v;
}

// ---------------------------------------------------------------------------
// Kernel 1: per-row zero-mean / unit-RMS normalization, folding the mask and
// 1/sqrt(L) so that sim = dot(v_a, v_b) directly. Output f16 into workspace.
// One wave32 per row (L=128 -> float4 per lane).
// ---------------------------------------------------------------------------
__global__ void rcg_normalize_f16(const float* __restrict__ history,
                                  const unsigned char* __restrict__ mask,
                                  _Float16* __restrict__ Xh, int rows) {
    const int lane = threadIdx.x & 31;
    const int wave = threadIdx.x >> 5;
    const int row  = blockIdx.x * 8 + wave;
    if (row >= rows) return;

    const float4 x = ((const float4*)(history + (size_t)row * LN))[lane];
    float mean = wave_sum32(x.x + x.y + x.z + x.w) * (1.0f / LN);
    float cx = x.x - mean, cy = x.y - mean, cz = x.z - mean, cw = x.w - mean;
    float ss = wave_sum32(cx * cx + cy * cy + cz * cz + cw * cw);
    float denom = fmaxf(sqrtf(ss * (1.0f / LN)), 1e-6f);
    float mval  = mask[row] ? 1.0f : 0.0f;
    float scale = mval / (denom * 11.313708498984760f);   // fold 1/sqrt(128)

    v4h o;
    o[0] = (_Float16)(cx * scale);
    o[1] = (_Float16)(cy * scale);
    o[2] = (_Float16)(cz * scale);
    o[3] = (_Float16)(cw * scale);
    ((v4h*)(Xh + (size_t)row * LN))[lane] = o;
}

// ---------------------------------------------------------------------------
// Kernel 2: fused WMMA GEMM (16 rows x 4096 cols, K=128) -> LDS, then per-row
// tournament top-20 + row-normalized sparse scatter. 256 threads = 8 waves;
// 16 * RSTR * 4 = 270,464 B dynamic LDS (CDNA5: up to 320 KB per workgroup).
// ---------------------------------------------------------------------------
__global__ void __launch_bounds__(256, 1)
rcg_corr_topk(const _Float16* __restrict__ Xh, float* __restrict__ out) {
    extern __shared__ float smem[];

    const int lane      = threadIdx.x & 31;
    const int wave      = threadIdx.x >> 5;
    const int tilesPerB = NN / 16;
    const int b         = blockIdx.x / tilesPerB;
    const int rowbase   = (blockIdx.x % tilesPerB) * 16;
    const int h         = lane >> 4;
    const int l15       = lane & 15;

    // --- Kick off the DRAM zero-fill of this wave's two output rows NOW so
    // the write-bound traffic drains behind the GEMM + top-k compute.
    const float4 z4 = make_float4(0.0f, 0.0f, 0.0f, 0.0f);
#pragma unroll
    for (int rr = 0; rr < 2; ++rr) {
        float* orow = out + ((size_t)b * NN + rowbase + wave * 2 + rr) * NN;
        for (int j = lane; j < NN / 4; j += 32)
            ((float4*)orow)[j] = z4;
    }

    const _Float16* Xb = Xh + (size_t)b * NN * LN;

    // --- A fragments: 16 rows, shared by all column tiles; 4 K-steps of 32.
    // A loaded k-contiguous here (B matches): lanes 0-15 hold K={0..15}, lanes 16-31 K={16..31}.
    const _Float16* rp = Xb + (size_t)(rowbase + l15) * LN;
    v16h afrag[4];
#pragma unroll
    for (int k = 0; k < 4; ++k) {
        const int kb = k * 32;
        v8h lo = *(const v8h*)(rp + kb + h * 16);
        v8h hi = *(const v8h*)(rp + kb + h * 16 + 8);
        afrag[k] = __builtin_shufflevector(lo, hi,
                     0, 1, 2, 3, 4, 5, 6, 7, 8, 9, 10, 11, 12, 13, 14, 15);
    }

    // --- Double-buffered GEMM over column tiles. B fragment lane n holds
    // X[col_n][k..k+15] contiguously: one 32-byte load per K-step.
    auto loadB = [&](int ct, v16h* dst) {
        const _Float16* cp = Xb + (size_t)(ct * 16 + l15) * LN;
#pragma unroll
        for (int k = 0; k < 4; ++k)
            dst[k] = *(const v16h*)(cp + k * 32 + h * 16);
    };
    auto computeStore = [&](int ct, const v16h* bf) {
        v8f c = {};
#pragma unroll
        for (int k = 0; k < 4; ++k)
            c = __builtin_amdgcn_wmma_f32_16x16x32_f16(
                    false, afrag[k], false, bf[k], (short)0, c, false, false);
        // col j -> block j>>5 (33-float pitch), elem j&31
        const int cbase = (ct >> 1) * 33 + (ct & 1) * 16 + l15;
#pragma unroll
        for (int v = 0; v < 8; ++v)
            smem[(v + (h << 3)) * RSTR + cbase] = fmaxf(c[v], 0.0f);
    };

    v16h ba[4], bb[4];
    loadB(wave, ba);
    for (int ct = wave; ct < tilesPerB; ct += 16) {
        loadB(ct + 8, bb);                       // prefetch tile ct+8
        computeStore(ct, ba);
        const int ctn = (ct + 16 < tilesPerB) ? ct + 16 : wave;  // branchless tail
        loadB(ctn, ba);                          // prefetch tile ct+16
        computeStore(ct + 8, bb);
    }
    __syncthreads();

    // --- Zero the 16 diagonal entries (branch-free hot loop above).
    if (threadIdx.x < 16) {
        const int g = rowbase + threadIdx.x;
        smem[threadIdx.x * RSTR + (g >> 5) * 33 + (g & 31)] = 0.0f;
    }
    __syncthreads();

    // --- Tournament top-20 per row; each wave owns rows 2w and 2w+1.
#pragma unroll
    for (int rr = 0; rr < 2; ++rr) {
        const int m = wave * 2 + rr;
        volatile float* row = smem + m * RSTR;

        // Build per-lane block maxima: lane owns blocks {lane, lane+32, +64, +96};
        // addresses 33*blk + t are bank-conflict-free across lanes.
        float lv[4];
#pragma unroll
        for (int s = 0; s < 4; ++s) {
            const int blk = lane + 32 * s;
            float mx = -3.0e38f;
            for (int t = 0; t < 32; ++t) mx = fmaxf(mx, row[33 * blk + t]);
            lv[s] = mx;
        }

        float topv[TOPK];
        int   topi[TOPK];
        float sum = 0.0f;
#pragma unroll
        for (int t = 0; t < TOPK; ++t) {
            // lane candidate: best of its 4 blocks (lowest block on tie)
            float bv = lv[0];
            int   bs = 0;
#pragma unroll
            for (int s = 1; s < 4; ++s)
                if (lv[s] > bv) { bv = lv[s]; bs = s; }
            int bblk = lane + 32 * bs;
            // wave argmax over 128 block maxima (value desc, block asc)
#pragma unroll
            for (int o = 16; o > 0; o >>= 1) {
                float ov = __shfl_xor(bv, o, 32);
                int   ob = __shfl_xor(bblk, o, 32);
                if (ov > bv || (ov == bv && ob < bblk)) { bv = ov; bblk = ob; }
            }
            // rescan winning block: one element per lane (consecutive banks)
            const float x   = row[33 * bblk + lane];
            const int   col = bblk * 32 + lane;
            float xv = x;
            int   xc = col;
#pragma unroll
            for (int o = 16; o > 0; o >>= 1) {
                float ov = __shfl_xor(xv, o, 32);
                int   oc = __shfl_xor(xc, o, 32);
                if (ov > xv || (ov == xv && oc < xc)) { xv = ov; xc = oc; }
            }
            topv[t] = xv;
            topi[t] = xc;
            sum += xv;
            // knockout winner (same-wave DS ordering covers future rescans)
            if (lane == (xc & 31)) row[33 * bblk + lane] = -3.0e38f;
            // new block max excluding the winner, from registers
            float x2 = (col == xc) ? -3.0e38f : x;
#pragma unroll
            for (int o = 16; o > 0; o >>= 1) x2 = fmaxf(x2, __shfl_xor(x2, o, 32));
            if (lane == (bblk & 31)) {
#pragma unroll
                for (int s = 0; s < 4; ++s)
                    if ((bblk >> 5) == s) lv[s] = x2;
            }
        }

        // scatter normalized values (stores ordered after this wave's zeros)
        const float scale = 1.0f / fmaxf(sum, 1e-6f);
        float* orow = out + ((size_t)b * NN + rowbase + m) * NN;
        if (lane == 0) {
#pragma unroll
            for (int t = 0; t < TOPK; ++t)
                orow[topi[t]] = topv[t] * scale;
        }
    }
}

extern "C" void kernel_launch(void* const* d_in, const int* in_sizes, int n_in,
                              void* d_out, int out_size, void* d_ws, size_t ws_size,
                              hipStream_t stream) {
    const float*         history = (const float*)d_in[0];
    const unsigned char* mask    = (const unsigned char*)d_in[1];
    const int rows = in_sizes[1];                   // B * N
    _Float16* Xh = (_Float16*)d_ws;                 // rows * 128 f16 = 16 MB scratch

    rcg_normalize_f16<<<(rows + 7) / 8, 256, 0, stream>>>(history, mask, Xh, rows);
    rcg_corr_topk<<<rows / 16, 256, 16 * RSTR * sizeof(float), stream>>>(
        Xh, (float*)d_out);
}
